// LocalAttention_61392262529469
// MI455X (gfx1250) — matrix operand order repaired and checked by reading, not compiled, not run on hardware
//
#include <hip/hip_runtime.h>
#include <hip/hip_bf16.h>
#include <math.h>

// Problem constants (from reference setup)
#define BB   32
#define TT   128
#define SS   1024
#define DIMM 1024
#define DWIN 10
#define MM   (BB * TT)      // 4096 rows
#define MAXW 24             // max window span (<=23)

typedef float     v2f  __attribute__((ext_vector_type(2)));
typedef float     v8f  __attribute__((ext_vector_type(8)));
typedef _Float16  v8h  __attribute__((ext_vector_type(8)));
typedef _Float16  v16h __attribute__((ext_vector_type(16)));
typedef _Float16  h4   __attribute__((ext_vector_type(4)));

#define LDA 1028            // f32 LDS tile row stride (floats): 4-bank shift/row
#define LDH 1032            // f16 LDS tile row stride (halves): 516 dwords = 4-bank shift/row

static __device__ __forceinline__ v16h cat8(v8h a, v8h b) {
    return __builtin_shufflevector(a, b, 0,1,2,3,4,5,6,7,8,9,10,11,12,13,14,15);
}

// ---------------------------------------------------------------------------
// Kernel 0: split W_out (f32) into hi/lo f16 arrays (same row-major layout)
// ---------------------------------------------------------------------------
__global__ void split_kernel(const float* __restrict__ src,
                             _Float16* __restrict__ hi,
                             _Float16* __restrict__ lo, int n4)
{
    const int i = blockIdx.x * 256 + threadIdx.x;
    if (i >= n4) return;
    const float4 v = ((const float4*)src)[i];
    h4 h, l;
    h.x = (_Float16)v.x; l.x = (_Float16)(v.x - (float)h.x);
    h.y = (_Float16)v.y; l.y = (_Float16)(v.y - (float)h.y);
    h.z = (_Float16)v.z; l.z = (_Float16)(v.z - (float)h.z);
    h.w = (_Float16)v.w; l.w = (_Float16)(v.w - (float)h.w);
    ((h4*)hi)[i] = h;
    ((h4*)lo)[i] = l;
}

// ---------------------------------------------------------------------------
// Kernel 1: p_t = (len-1) * sigmoid( v_pred . tanh(input @ W_pred^T) )
// Full-fp32 WMMA (16x16x4) — p_t feeds hard integer window thresholds.
// ---------------------------------------------------------------------------
__global__ __launch_bounds__(256, 1)
void pred_kernel(const float* __restrict__ inp,
                 const float* __restrict__ Wp,   // [1024][1024] row = out feature
                 const float* __restrict__ vp,   // [1024]
                 const int*   __restrict__ lens, // [32]
                 float*       __restrict__ p_out) // [4096]
{
    extern __shared__ float atile[];      // 16 x LDA floats
    __shared__ float red[8][16];

    const int tid  = threadIdx.x;
    const int wave = tid >> 5;
    const int lane = tid & 31;
    const int m0   = blockIdx.x * 16;

    for (int i = tid; i < 16 * 256; i += 256) {
        const int row = i >> 8;
        const int c4  = (i & 255) << 2;
        *(float4*)(atile + row * LDA + c4) =
            *(const float4*)(inp + (size_t)(m0 + row) * DIMM + c4);
    }
    __syncthreads();

    const int hl = lane & 15;
    const int kh = (lane >> 4) << 1;      // K sub-offset: 0 or 2
    const float* arow = atile + hl * LDA;

    v8f acc[8] = {};
    for (int kk = 0; kk < DIMM; kk += 4) {
        v2f a = *(const v2f*)(arow + kk + kh);
        #pragma unroll
        for (int j = 0; j < 8; ++j) {
            const int n = (wave * 8 + j) * 16 + hl;
            v2f b = *(const v2f*)(Wp + (size_t)n * DIMM + kk + kh);
            acc[j] = __builtin_amdgcn_wmma_f32_16x16x4_f32(
                false, a, false, b, (short)0, acc[j], false, false);
        }
    }

    float pacc[8];
    #pragma unroll
    for (int r = 0; r < 8; ++r) pacc[r] = 0.f;
    #pragma unroll
    for (int j = 0; j < 8; ++j) {
        const int n = (wave * 8 + j) * 16 + hl;
        const float v = vp[n];
        #pragma unroll
        for (int r = 0; r < 8; ++r) pacc[r] += tanhf(acc[j][r]) * v;
    }
    #pragma unroll
    for (int r = 0; r < 8; ++r) {
        float v = pacc[r];
        v += __shfl_xor(v, 1, 16);
        v += __shfl_xor(v, 2, 16);
        v += __shfl_xor(v, 4, 16);
        v += __shfl_xor(v, 8, 16);
        if (lane == 0)  red[wave][r]     = v;   // rows 0..7
        if (lane == 16) red[wave][8 + r] = v;   // rows 8..15
    }
    __syncthreads();
    if (tid < 16) {
        float s = 0.f;
        #pragma unroll
        for (int w = 0; w < 8; ++w) s += red[w][tid];
        const int m = m0 + tid;
        const int b = m >> 7;                  // T = 128
        const float lf = (float)lens[b];
        const float sig = 1.f / (1.f + expf(-s));
        p_out[m] = (lf - 1.f) * sig;
    }
}

// ---------------------------------------------------------------------------
// Kernel 2: per-(b,t) windowed scores, softmax-with-zero-background, Gaussian
// reweighting; writes the full a row ([T,B,S]) and context c (f32).
// ---------------------------------------------------------------------------
__global__ void window_kernel(const float* __restrict__ inp,
                              const float* __restrict__ mem,
                              const int*   __restrict__ lens,
                              const float* __restrict__ p_in,
                              float*       __restrict__ a_out, // [T,B,S]
                              float*       __restrict__ c_out) // [4096,1024]
{
    __shared__ float scores[MAXW];
    __shared__ int   sval[MAXW];
    __shared__ float wv[MAXW];

    const int m    = blockIdx.x;
    const int b    = m >> 7;
    const int t    = m & 127;
    const int tid  = threadIdx.x;
    const int wave = tid >> 5;
    const int lane = tid & 31;

    const float p   = p_in[m];
    const int   len = lens[b];
    const float lf  = (float)len;

    int lo = (int)floorf(p - (float)DWIN); if (lo < 0) lo = 0;
    int hi = (int)ceilf(p + (float)DWIN);
    if (hi > len - 1) hi = len - 1;
    if (hi > SS - 1)  hi = SS - 1;

    if (tid < MAXW) { scores[tid] = 0.f; sval[tid] = 0; wv[tid] = 0.f; }
    __syncthreads();

    const float* irow = inp + (size_t)m * DIMM;
    for (int jr = wave; jr < MAXW; jr += 8) {
        const int s = lo + jr;
        if (s > hi) continue;
        const float fs = (float)s;
        const bool pass = (fs >= p - (float)DWIN) && (fs <= p + (float)DWIN) &&
                          (fs <= lf - 1.f);
        if (!pass) continue;
        const float* mrow = mem + ((size_t)b * SS + s) * DIMM;
        float sum = 0.f;
        #pragma unroll
        for (int k2 = 0; k2 < 8; ++k2) {
            const int off = lane * 4 + k2 * 128;
            const float4 x = *(const float4*)(irow + off);
            const float4 y = *(const float4*)(mrow + off);
            sum += x.x * y.x + x.y * y.y + x.z * y.z + x.w * y.w;
        }
        sum += __shfl_xor(sum, 1,  32);
        sum += __shfl_xor(sum, 2,  32);
        sum += __shfl_xor(sum, 4,  32);
        sum += __shfl_xor(sum, 8,  32);
        sum += __shfl_xor(sum, 16, 32);
        if (lane == 0) { scores[jr] = sum; sval[jr] = 1; }
    }
    __syncthreads();

    if (tid == 0) {
        int cnt = 0;
        float mx = 0.f;                      // len >= 512 >> window, so score 0 exists
        for (int jr = 0; jr < MAXW; ++jr)
            if (sval[jr]) { ++cnt; mx = fmaxf(mx, scores[jr]); }
        float denom = (lf - (float)cnt) * expf(-mx);
        for (int jr = 0; jr < MAXW; ++jr)
            if (sval[jr]) denom += expf(scores[jr] - mx);
        for (int jr = 0; jr < MAXW; ++jr) {
            if (sval[jr]) {
                const float fs = (float)(lo + jr);
                const float d  = fs - p;
                const float g  = expf(-(d * d) * 0.02f);   // 2*(D/2)^2 = 50
                wv[jr] = expf(scores[jr] - mx) / denom * g;
            }
        }
    }
    __syncthreads();

    {
        float* arow = a_out + ((size_t)t * BB + b) * SS;
        const int s0 = tid * 4;
        float4 o;
        float* op = (float*)&o;
        #pragma unroll
        for (int q = 0; q < 4; ++q) {
            const int jr = s0 + q - lo;
            op[q] = (jr >= 0 && jr < MAXW) ? wv[jr] : 0.f;
        }
        *(float4*)(arow + s0) = o;
    }

    {
        const int d0 = tid * 4;
        float4 cacc = {0.f, 0.f, 0.f, 0.f};
        for (int jr = 0; jr < MAXW; ++jr) {
            const float w = wv[jr];
            if (w != 0.f) {
                const float4 mv =
                    *(const float4*)(mem + ((size_t)b * SS + (lo + jr)) * DIMM + d0);
                cacc.x += w * mv.x; cacc.y += w * mv.y;
                cacc.z += w * mv.z; cacc.w += w * mv.w;
            }
        }
        *(float4*)(c_out + (size_t)m * DIMM + d0) = cacc;
    }
}

// ---------------------------------------------------------------------------
// Kernel 3: attn_h = tanh([c | input] @ W_out^T), stored [T,B,dim].
// Split-f16 WMMA (hi*hi + hi*lo + lo*hi on v_wmma_f32_16x16x32_f16):
// ~2^-22 relative error at ~2.7x fewer matrix issues than f32 16x16x4.
// A tiles split f32 -> (hi,lo) f16 while staging into LDS.
// ---------------------------------------------------------------------------
__global__ __launch_bounds__(256, 1)
void outproj_kernel(const float*    __restrict__ c_in, // [4096,1024] f32
                    const float*    __restrict__ inp,  // [4096,1024] f32
                    const _Float16* __restrict__ woh,  // [1024][2048] hi
                    const _Float16* __restrict__ wol,  // [1024][2048] lo
                    float*          __restrict__ h_out) // [T,B,1024]
{
    extern __shared__ _Float16 hsm[];     // ahi[16*LDH] ++ alo[16*LDH]
    const int tid  = threadIdx.x;
    const int wave = tid >> 5;
    const int lane = tid & 31;
    const int m0   = blockIdx.x * 16;
    const int hl   = lane & 15;
    const int g    = lane >> 4;           // K half-group select

    _Float16* ahi = hsm;
    _Float16* alo = hsm + 16 * LDH;

    v8f acc[8] = {};

    for (int phase = 0; phase < 2; ++phase) {
        const float* src = phase ? inp : c_in;
        __syncthreads();
        // stage + split 16x1024 f32 tile into hi/lo f16 LDS tiles
        for (int i = tid; i < 16 * 256; i += 256) {
            const int row = i >> 8;
            const int c4  = (i & 255) << 2;
            const float4 v = *(const float4*)(src + (size_t)(m0 + row) * DIMM + c4);
            h4 h, l;
            h.x = (_Float16)v.x; l.x = (_Float16)(v.x - (float)h.x);
            h.y = (_Float16)v.y; l.y = (_Float16)(v.y - (float)h.y);
            h.z = (_Float16)v.z; l.z = (_Float16)(v.z - (float)h.z);
            h.w = (_Float16)v.w; l.w = (_Float16)(v.w - (float)h.w);
            *(h4*)(ahi + row * LDH + c4) = h;
            *(h4*)(alo + row * LDH + c4) = l;
        }
        __syncthreads();

        const _Float16* arh = ahi + hl * LDH;
        const _Float16* arl = alo + hl * LDH;
        const int kbase = phase * DIMM;

        for (int kb = 0; kb < DIMM; kb += 32) {
            // A fragment: 16 contiguous K halves at K = kb + 16*g
            const v16h Ah = cat8(*(const v8h*)(arh + kb + 16 * g),
                                 *(const v8h*)(arh + kb + 16 * g + 8));
            const v16h Al = cat8(*(const v8h*)(arl + kb + 16 * g),
                                 *(const v8h*)(arl + kb + 16 * g + 8));
            #pragma unroll
            for (int j = 0; j < 8; ++j) {
                const int n = (wave * 8 + j) * 16 + hl;
                // B fragment: 16 contiguous K halves at K = kb + 16*g
                const size_t wo = (size_t)n * (2 * DIMM) + kbase + kb + 16 * g;
                const v16h Bh = cat8(*(const v8h*)(woh + wo),
                                     *(const v8h*)(woh + wo + 8));
                const v16h Bl = cat8(*(const v8h*)(wol + wo),
                                     *(const v8h*)(wol + wo + 8));
                acc[j] = __builtin_amdgcn_wmma_f32_16x16x32_f16(
                    false, Ah, false, Bh, (short)0, acc[j], false, false);
                acc[j] = __builtin_amdgcn_wmma_f32_16x16x32_f16(
                    false, Ah, false, Bl, (short)0, acc[j], false, false);
                acc[j] = __builtin_amdgcn_wmma_f32_16x16x32_f16(
                    false, Al, false, Bh, (short)0, acc[j], false, false);
            }
        }
    }

    // tanh epilogue + transposed store: row m -> out[t][b][n]
    const int rbase = (lane < 16) ? 0 : 8;
    #pragma unroll
    for (int j = 0; j < 8; ++j) {
        const int n = (wave * 8 + j) * 16 + hl;
        #pragma unroll
        for (int r = 0; r < 8; ++r) {
            const int m  = m0 + rbase + r;
            const int bb = m >> 7;
            const int t  = m & 127;
            h_out[((size_t)t * BB + bb) * DIMM + n] = tanhf(acc[j][r]);
        }
    }
}

// ---------------------------------------------------------------------------
extern "C" void kernel_launch(void* const* d_in, const int* in_sizes, int n_in,
                              void* d_out, int out_size, void* d_ws, size_t ws_size,
                              hipStream_t stream)
{
    (void)in_sizes; (void)n_in; (void)out_size; (void)ws_size;

    const float* inp  = (const float*)d_in[0]; // [B,T,dim]
    const float* mem  = (const float*)d_in[1]; // [B,S,dim]
    const int*   lens = (const int*)  d_in[2]; // [B]
    const float* Wo   = (const float*)d_in[3]; // [dim, 2*dim]
    const float* Wp   = (const float*)d_in[4]; // [dim, dim]
    const float* vp   = (const float*)d_in[5]; // [1, dim]

    float* attn = (float*)d_out;                           // [T,B,dim]
    float* aout = attn + (size_t)TT * BB * DIMM;           // [T,B,S]

    // workspace: p_t [4096] f32 | c [4096x1024] f32 | W_out hi/lo f16
    float* p_ws = (float*)d_ws;
    float* c_ws = p_ws + MM;
    _Float16* woh = (_Float16*)(c_ws + (size_t)MM * DIMM);
    _Float16* wol = woh + (size_t)DIMM * (2 * DIMM);

    const int wo_n4 = (DIMM * 2 * DIMM) / 4;               // 524288 float4s

    split_kernel  <<<wo_n4 / 256, 256, 0, stream>>>(Wo, woh, wol, wo_n4);
    pred_kernel   <<<MM / 16, 256, (size_t)16 * LDA * sizeof(float), stream>>>
                   (inp, Wp, vp, lens, p_ws);
    window_kernel <<<MM, 256, 0, stream>>>(inp, mem, lens, p_ws, aout, c_ws);
    outproj_kernel<<<MM / 16, 256, (size_t)2 * 16 * LDH * sizeof(_Float16), stream>>>
                   (c_ws, inp, woh, wol, attn);
}
